// ElmanRNN_13185549599212
// MI455X (gfx1250) — matrix operand order repaired and checked by reading, not compiled, not run on hardware
//
#include <hip/hip_runtime.h>

// Elman RNN fused scan for gfx1250 (MI455X), wave32 + WMMA bf16.
//   x:[128,2048,128] W_in:[256,128] b_in:[256] W_rec:[256,256] W_out:[128,256] b_out:[128]
//   out:[128,128] = (scan_t sigmoid(b_in + x_t@W_in^T + h@W_rec^T)) @ W_out^T + b_out
//
// 8 persistent workgroups (one 16-batch-row tile each; recurrence is independent
// across batch rows -> no inter-WG sync). 16 waves/WG, each wave owns one 16-col
// hidden tile. Combined-K GEMM per step: A=[h | x_t] (16x384 bf16 in LDS, double
// buffered), B=[W_rec^T ; W_in^T] held in VGPRs for the whole kernel.
// sched_barrier forces: 24 ds_load_b128 up front, then 12 back-to-back WMMAs,
// so the accumulation chain issues at pipe rate instead of LDS latency.

#define BATCH 128
#define SEQ   2048
#define INP   128
#define HID   256
#define ONUM  128
#define RS    392   // LDS row stride in bf16 elems: 384 data + 8 pad (kills ds bank conflicts)

typedef __attribute__((ext_vector_type(16))) __bf16 v16bf;
typedef __attribute__((ext_vector_type(8)))  __bf16 v8bf;
typedef __attribute__((ext_vector_type(4)))  __bf16 v4bf;
typedef __attribute__((ext_vector_type(8)))  float  v8f;

// A operand (16-bit 16x32, ISA 7.12.2): lane<16 holds K {0..7,16..23}, lane>=16
// holds K {8..15,24..31}. p points at row base + kb*32 + (lane>>4)*8.
__device__ __forceinline__ v16bf load_a_lds(const __bf16* p) {
  v8bf lo = *(const v8bf*)(p);        // ds_load_b128
  v8bf hi = *(const v8bf*)(p + 16);   // ds_load_b128
  v16bf r;
#pragma unroll
  for (int j = 0; j < 8; ++j) { r[j] = lo[j]; r[j + 8] = hi[j]; }
  return r;
}

// B operand (16-bit 32x16): lane holds K {0..7,16..23} (lane<16) or {8..15,24..31},
// for column n = lane&15. Source is an f32 weight row; convert to bf16.
__device__ __forceinline__ v16bf load_b_f32(const float* p) {
  v16bf r;
#pragma unroll
  for (int j = 0; j < 16; ++j) r[j] = (__bf16)p[j < 8 ? j : j + 8];
  return r;
}

__device__ __forceinline__ float sigmoid_fast(float z) {
  float e = __builtin_amdgcn_exp2f(-1.4426950408889634f * z); // v_exp_f32 (trans)
  return __builtin_amdgcn_rcpf(1.0f + e);                     // v_rcp_f32 (trans)
}

__global__ __launch_bounds__(512)
void elman_fused(const float* __restrict__ x,     const float* __restrict__ W_in,
                 const float* __restrict__ b_in,  const float* __restrict__ W_rec,
                 const float* __restrict__ W_out, const float* __restrict__ b_out,
                 float* __restrict__ out) {
  __shared__ __bf16 abuf[2][16 * RS];   // double-buffered [h(256) | x_t(128)] rows=batch tile

  const int tid  = threadIdx.x;
  const int lane = tid & 31;
  const int wave = tid >> 5;        // 0..15
  const int m16  = lane & 15;       // A row / B,C column within tile
  const int kg   = lane >> 4;       // K half-group select
  const int aoff = kg * 8;          // A lane K offset
  const int boff = kg * 8;          // B lane K offset
  const int mrow = kg << 3;         // C/D row base: r (lanes 0-15) or r+8 (lanes 16-31)
  const int b0   = blockIdx.x * 16; // batch tile base
  const int n0   = wave * 16;       // hidden tile base (16 waves * 16 = 256)

  // ---- one-time: combined B = [W_rec^T (K=0..255) ; W_in^T (K=256..383)] in VGPRs ----
  v16bf bmat[12];
#pragma unroll
  for (int kb = 0; kb < 8; ++kb)
    bmat[kb] = load_b_f32(W_rec + (size_t)(n0 + m16) * HID + kb * 32 + boff);
#pragma unroll
  for (int kb = 0; kb < 4; ++kb)
    bmat[8 + kb] = load_b_f32(W_in + (size_t)(n0 + m16) * INP + kb * 32 + boff);
  const float bias = b_in[n0 + m16];

  // ---- init buf0: h0 = 0 (cols 0..255), stage x[t=0] (cols 256..383) ----
  const int xr = wave;          // row 0..15 of the tile
  const int xc = lane * 4;      // f32 column group 0..124
  {
    uint4 z = make_uint4(0u, 0u, 0u, 0u);
    *(uint4*)&abuf[0][xr * RS + lane * 8] = z;  // 8 bf16 zeros -> covers cols 0..255
    float4 v = *(const float4*)(x + ((size_t)(b0 + xr) * SEQ + 0) * INP + xc);
    v4bf xv; xv[0] = (__bf16)v.x; xv[1] = (__bf16)v.y; xv[2] = (__bf16)v.z; xv[3] = (__bf16)v.w;
    *(v4bf*)&abuf[0][xr * RS + 256 + xc] = xv;
  }
  __syncthreads();

  // ---- sequential scan: one __syncthreads per step, everything else on-WGP ----
  for (int t = 0; t < SEQ; ++t) {
    const __bf16* cur = abuf[t & 1];
    __bf16*       nxt = abuf[(t + 1) & 1];

    // prefetch x[t+1] (overlaps the WMMA chain below)
    float4 xp = make_float4(0.f, 0.f, 0.f, 0.f);
    if (t + 1 < SEQ)
      xp = *(const float4*)(x + ((size_t)(b0 + xr) * SEQ + (t + 1)) * INP + xc);

    // Preload ALL A-tiles for this step (24 back-to-back ds_load_b128).
    v16bf a[12];
#pragma unroll
    for (int kb = 0; kb < 12; ++kb)
      a[kb] = load_a_lds(cur + m16 * RS + kb * 32 + aoff);

    // Fence the machine scheduler: keep every ds_load above, every WMMA below,
    // so only one s_wait_dscnt is paid and the 12 WMMAs issue back-to-back.
    __builtin_amdgcn_sched_barrier(0);

    // C = b_in broadcast
    v8f c;
#pragma unroll
    for (int r = 0; r < 8; ++r) c[r] = bias;

    // z = b_in + [h | x_t] @ [W_rec^T ; W_in^T]  (K = 384, 12 WMMAs, D->C pipelined)
#pragma unroll
    for (int kb = 0; kb < 12; ++kb)
      c = __builtin_amdgcn_wmma_f32_16x16x32_bf16(false, a[kb], false, bmat[kb],
                                                  (short)0, c, false, false);

    // h_next = sigmoid(z) -> bf16 into next buffer (this wave's 16-col slice)
#pragma unroll
    for (int r = 0; r < 8; ++r) {
      float s = sigmoid_fast(c[r]);
      nxt[(mrow + r) * RS + n0 + m16] = (__bf16)s;
    }

    // commit prefetched x[t+1] into next buffer
    if (t + 1 < SEQ) {
      v4bf xv; xv[0] = (__bf16)xp.x; xv[1] = (__bf16)xp.y; xv[2] = (__bf16)xp.z; xv[3] = (__bf16)xp.w;
      *(v4bf*)&nxt[xr * RS + 256 + xc] = xv;
    }
    __syncthreads();
  }

  // ---- out = h_final @ W_out^T + b_out ; h_final lives in abuf[0] (SEQ even) ----
  if (wave < 8) {                       // O = 128 -> 8 tiles; wave-uniform branch, EXEC full
    const int o0 = wave * 16;
    const float bo = b_out[o0 + m16];
    v8f c;
#pragma unroll
    for (int r = 0; r < 8; ++r) c[r] = bo;
#pragma unroll
    for (int kb = 0; kb < 8; ++kb) {
      v16bf a = load_a_lds(abuf[0] + m16 * RS + kb * 32 + aoff);
      v16bf b = load_b_f32(W_out + (size_t)(o0 + m16) * HID + kb * 32 + boff);
      c = __builtin_amdgcn_wmma_f32_16x16x32_bf16(false, a, false, b,
                                                  (short)0, c, false, false);
    }
#pragma unroll
    for (int r = 0; r < 8; ++r)
      out[(size_t)(b0 + mrow + r) * ONUM + o0 + m16] = c[r];
  }
}

extern "C" void kernel_launch(void* const* d_in, const int* in_sizes, int n_in,
                              void* d_out, int out_size, void* d_ws, size_t ws_size,
                              hipStream_t stream) {
  (void)in_sizes; (void)n_in; (void)out_size; (void)d_ws; (void)ws_size;
  const float* x     = (const float*)d_in[0];
  const float* W_in  = (const float*)d_in[1];
  const float* b_in  = (const float*)d_in[2];
  const float* W_rec = (const float*)d_in[3];
  const float* W_out = (const float*)d_in[4];
  const float* b_out = (const float*)d_in[5];
  float* out = (float*)d_out;

  elman_fused<<<dim3(BATCH / 16), dim3(512), 0, stream>>>(
      x, W_in, b_in, W_rec, W_out, b_out, out);
}
